// Model_30992484008577
// MI455X (gfx1250) — 8-bit matrix operand order repaired and checked by reading, compiled, not run on hardware
//
#include <hip/hip_runtime.h>


typedef __attribute__((ext_vector_type(16))) int   v16i;
typedef __attribute__((ext_vector_type(8)))  float v8f;

#define DIMS   1000
#define NPAD   1024
#define NT     64
#define PPIX   784
#define LVL    10
#define KLOG   (PPIX*LVL)
#define KPAD   7936
#define KT     62
#define BATCH  256
#define MT     16
#define NCLS   10

#define FP8_ONE  0x38u
#define FP8_NEG1 0xB8u

#define AFRAG_BYTES ((size_t)MT * KT * 32 * 64)
#define BFRAG_BYTES ((size_t)NT * KT * 32 * 64)

__global__ void build_a(const float* __restrict__ x, unsigned char* __restrict__ afrag) {
    unsigned int t = blockIdx.x * blockDim.x + threadIdx.x;
    unsigned int word = t & 15;
    unsigned int lane = (t >> 4) & 31;
    unsigned int kt   = (t >> 9) % KT;
    unsigned int mt   = (t >> 9) / KT;
    if (mt >= MT) return;
    unsigned int m = mt * 16 + (lane & 15);
    unsigned int out = 0;
    #pragma unroll
    for (int j = 0; j < 4; ++j) {
        unsigned int byte = word * 4 + j;
        unsigned int half = byte >> 5;
        unsigned int c    = (byte >> 3) & 3;
        unsigned int b0   = byte & 7;
        unsigned int k64  = c * 16 + ((lane >> 4) & 1) * 8 + b0;
        unsigned int k    = kt * 128 + half * 64 + k64;
        unsigned int val  = 0;
        if (k < KLOG) {
            unsigned int p = k / LVL, l = k % LVL;
            int idx = (int)rintf(x[m * PPIX + p] * (float)(LVL - 1));
            idx = idx < 0 ? 0 : (idx > LVL - 1 ? LVL - 1 : idx);
            if ((unsigned int)idx == l) val = FP8_ONE;
        }
        out |= val << (8 * j);
    }
    ((unsigned int*)afrag)[t] = out;
}

__global__ void build_b(const float* __restrict__ pos, const float* __restrict__ vt,
                        unsigned char* __restrict__ bfrag) {
    unsigned int t = blockIdx.x * blockDim.x + threadIdx.x;
    unsigned int word = t & 15;
    unsigned int lane = (t >> 4) & 31;
    unsigned int kt   = (t >> 9) % KT;
    unsigned int nt   = (t >> 9) / KT;
    if (nt >= NT) return;
    unsigned int d = nt * 16 + (lane & 15);
    unsigned int out = 0;
    if (d < DIMS) {
        #pragma unroll
        for (int j = 0; j < 4; ++j) {
            unsigned int byte = word * 4 + j;
            unsigned int kk = ((byte >> 5) << 6) | (((byte >> 3) & 3) << 4) | (((lane >> 4) & 1) << 3) | (byte & 7);
            unsigned int k  = kt * 128 + kk;
            unsigned int val = 0;
            if (k < KLOG) {
                unsigned int p = k / LVL, l = k % LVL;
                float s = pos[p * DIMS + d] * vt[l * DIMS + d];
                val = (s > 0.f) ? FP8_ONE : FP8_NEG1;
            }
            out |= val << (8 * j);
        }
    }
    ((unsigned int*)bfrag)[t] = out;
}

__global__ void __launch_bounds__(32) hdc_gemm(const unsigned char* __restrict__ afrag,
                                               const unsigned char* __restrict__ bfrag,
                                               float* __restrict__ enc) {
    const unsigned int lane = threadIdx.x;
    const unsigned int mt   = blockIdx.x >> 5;
    const unsigned int nt0  = (blockIdx.x & 31) * 2;

    v8f acc0 = {}, acc1 = {};

    const v16i* ap  = (const v16i*)(afrag + ((size_t)mt * KT * 32 + lane) * 64);
    const v16i* b0p = (const v16i*)(bfrag + ((size_t)(nt0 + 0) * KT * 32 + lane) * 64);
    const v16i* b1p = (const v16i*)(bfrag + ((size_t)(nt0 + 1) * KT * 32 + lane) * 64);

    v16i a  = ap[0];
    v16i b0 = b0p[0];
    v16i b1 = b1p[0];

    for (int kt = 0; kt < KT - 1; ++kt) {
        const size_t idx = (size_t)(kt + 1) * 32;
        v16i an  = ap[idx];
        v16i bn0 = b0p[idx];
        v16i bn1 = b1p[idx];
        acc0 = __builtin_amdgcn_wmma_f32_16x16x128_fp8_fp8(a, b0, (short)0, acc0, false, false);
        acc1 = __builtin_amdgcn_wmma_f32_16x16x128_fp8_fp8(a, b1, (short)0, acc1, false, false);
        a = an; b0 = bn0; b1 = bn1;
    }
    acc0 = __builtin_amdgcn_wmma_f32_16x16x128_fp8_fp8(a, b0, (short)0, acc0, false, false);
    acc1 = __builtin_amdgcn_wmma_f32_16x16x128_fp8_fp8(a, b1, (short)0, acc1, false, false);

    const unsigned int rowbase = mt * 16 + ((lane >> 4) << 3);
    const unsigned int nlo = lane & 15;
    {
        union { v8f v; float f[8]; } u;
        u.v = acc0;
        const unsigned int col = (nt0 + 0) * 16 + nlo;
        #pragma unroll
        for (int r = 0; r < 8; ++r)
            enc[(size_t)(rowbase + r) * NPAD + col] = (u.f[r] > 0.f) ? 1.f : -1.f;
    }
    {
        union { v8f v; float f[8]; } u;
        u.v = acc1;
        const unsigned int col = (nt0 + 1) * 16 + nlo;
        #pragma unroll
        for (int r = 0; r < 8; ++r)
            enc[(size_t)(rowbase + r) * NPAD + col] = (u.f[r] > 0.f) ? 1.f : -1.f;
    }
}

__global__ void __launch_bounds__(32) logits_kernel(const float* __restrict__ enc,
                                                    const float* __restrict__ W,
                                                    float* __restrict__ out) {
    const int b = blockIdx.x;
    const int lane = threadIdx.x;
    float acc[NCLS];
    #pragma unroll
    for (int c = 0; c < NCLS; ++c) acc[c] = 0.f;
    for (int d = lane; d < DIMS; d += 32) {
        const float e = enc[(size_t)b * NPAD + d];
        #pragma unroll
        for (int c = 0; c < NCLS; ++c) acc[c] += e * W[c * DIMS + d];
    }
    #pragma unroll
    for (int c = 0; c < NCLS; ++c) {
        float v = acc[c];
        #pragma unroll
        for (int off = 16; off > 0; off >>= 1) v += __shfl_xor(v, off, 32);
        if (lane == 0) out[b * NCLS + c] = v;
    }
}

extern "C" void kernel_launch(void* const* d_in, const int* in_sizes, int n_in,
                              void* d_out, int out_size, void* d_ws, size_t ws_size,
                              hipStream_t stream) {
    (void)in_sizes; (void)n_in; (void)out_size; (void)ws_size;
    const float* x   = (const float*)d_in[0];
    const float* pos = (const float*)d_in[1];
    const float* vt  = (const float*)d_in[2];
    const float* W   = (const float*)d_in[3];

    unsigned char* ws = (unsigned char*)d_ws;
    unsigned char* afrag = ws;
    unsigned char* bfrag = ws + AFRAG_BYTES;
    float*         enc   = (float*)(ws + AFRAG_BYTES + BFRAG_BYTES);

    {
        const unsigned int nthreads = MT * KT * 32 * 16;
        build_a<<<nthreads / 256, 256, 0, stream>>>(x, afrag);
    }
    {
        const unsigned int nthreads = NT * KT * 32 * 16;
        build_b<<<nthreads / 256, 256, 0, stream>>>(pos, vt, bfrag);
    }
    hdc_gemm<<<MT * 32, 32, 0, stream>>>(afrag, bfrag, enc);
    logits_kernel<<<BATCH, 32, 0, stream>>>(enc, W, (float*)d_out);
}
